// EGT_Attention_68178310857266
// MI455X (gfx1250) — matrix operand order repaired and checked by reading, not compiled, not run on hardware
//
#include <hip/hip_runtime.h>
#include <hip/hip_bf16.h>
#include <math.h>

typedef __bf16 bf16;
typedef __attribute__((ext_vector_type(8)))  __bf16 v8bf;
typedef __attribute__((ext_vector_type(16))) __bf16 v16bf;
typedef __attribute__((ext_vector_type(8)))  float  v8f;
typedef __attribute__((ext_vector_type(4)))  float  f4v;
typedef __attribute__((ext_vector_type(4)))  int    v4i;

#define DEVINL __device__ __forceinline__

// ---- optional CDNA5 async global->LDS path (compile-safe feature probe) ----
#if defined(__has_builtin)
#  if __has_builtin(__builtin_amdgcn_global_load_async_to_lds_b128)
#    define EGT_ASYNC_LDS 1
#  endif
#  if __has_builtin(__builtin_amdgcn_s_wait_asynccnt)
#    define EGT_WAIT_ASYNC_BUILTIN 1
#  endif
#endif

#ifdef EGT_ASYNC_LDS
typedef __attribute__((address_space(1))) v4i as1_v4i;
typedef __attribute__((address_space(3))) v4i as3_v4i;
// addr[31:0] of a generic LDS pointer is the LDS offset (ISA aperture rules);
// integer round-trip avoids generic->AS addrspacecast diagnostics.
DEVINL as3_v4i* lds_v4(void* p) { return (as3_v4i*)(unsigned)(size_t)p; }
DEVINL as1_v4i* glb_v4(const void* p) { return (as1_v4i*)(size_t)p; }
DEVINL void egt_wait_async() {
#  ifdef EGT_WAIT_ASYNC_BUILTIN
  __builtin_amdgcn_s_wait_asynccnt(0);
#  else
  asm volatile("s_wait_asynccnt 0x0" ::: "memory");
#  endif
}
#endif

// ---------------- problem constants ----------------
static constexpr int BB = 8, NN = 512, EW = 64, HH = 16, DD = 32;
static constexpr float QSCALE = 0.17677669529663687f; // 32^-0.5

// ---------------- workspace layout (bytes) ----------------
static constexpr size_t WS_WQKVT = 0;                                    // [1536][512] bf16
static constexpr size_t WS_WOHT  = WS_WQKVT + (size_t)1536*512*2;        // [512][512] bf16
static constexpr size_t WS_WEGT  = WS_WOHT  + (size_t)512*512*2;         // [32][64]  bf16
static constexpr size_t WS_WOET  = WS_WEGT  + (size_t)32*64*2;           // [64][16]  bf16
static constexpr size_t WS_QS    = (WS_WOET + (size_t)64*16*2 + 255) & ~(size_t)255; // [B][H][N][D] bf16
static constexpr size_t WS_KST   = WS_QS  + (size_t)BB*HH*NN*DD*2;       // [B][N][H][D] bf16
static constexpr size_t WS_VST   = WS_KST + (size_t)BB*NN*HH*DD*2;       // [B][H][D][N] bf16

// ---------------- dynamic LDS layout for egt_main (bytes) ----------------
static constexpr int OFF_ELN  = 0;                      // 256*64 bf16 (reused as Vatt[16][528] bf16 at end)
static constexpr int OFF_EBUF = 32768;                  // [16][260] f32  (E tile; aliased as Hf after QK)
static constexpr int OFF_GBUF = OFF_EBUF + 16*260*4;    // [16][260] f32  (G tile)
static constexpr int OFF_HB   = OFF_GBUF + 16*260*4;    // [256][16] bf16 (H_hat, A-operand for e_out)
static constexpr int OFF_KCH  = OFF_HB  + 256*16*2;     // [16][16][32] bf16
static constexpr int OFF_VCH  = OFF_KCH + 16*16*32*2;   // [16][32][16] bf16
static constexpr int OFF_AB   = OFF_VCH + 16*32*16*2;   // [16][272] bf16 (A_tild, padded)
static constexpr int OFF_MASK = OFF_AB  + 16*272*2;     // [256] f32
static constexpr int OFF_MROW = OFF_MASK + 1024;        // [256] f32
static constexpr int OFF_SROW = OFF_MROW + 1024;
static constexpr int OFF_DEG  = OFF_SROW + 1024;
static constexpr int OFF_FAC  = OFF_DEG  + 1024;
static constexpr int OFF_LNG  = OFF_FAC  + 1024;        // [64] f32
static constexpr int OFF_LNB  = OFF_LNG  + 256;         // [64] f32
static constexpr int SMEM_SZ  = OFF_LNB  + 256;         // = 121344 B (< 160 KB)

// ---------------- small helpers ----------------
DEVINL v16bf cat8(v8bf a, v8bf b) {
  return __builtin_shufflevector(a, b, 0,1,2,3,4,5,6,7,8,9,10,11,12,13,14,15);
}
DEVINL v8bf z8() {
  v8bf z;
#pragma unroll
  for (int i = 0; i < 8; ++i) z[i] = (bf16)0.0f;
  return z;
}
DEVINL v16bf z16() { return cat8(z8(), z8()); }
DEVINL v8f zf() {
  v8f z;
#pragma unroll
  for (int i = 0; i < 8; ++i) z[i] = 0.0f;
  return z;
}
DEVINL v8f wmma_bf16(v16bf a, v16bf b, v8f c) {
  // D(f32,16x16) = A(bf16,16x32) * B(bf16,32x16) + C
  return __builtin_amdgcn_wmma_f32_16x16x32_bf16(false, a, false, b, (short)0, c, false, false);
}

// =====================================================================
// Kernel 0: weight transpose + f32->bf16 conversion into B-frag layouts
// =====================================================================
__global__ void egt_prep(const float* __restrict__ Wqkv, const float* __restrict__ Woh,
                         const float* __restrict__ Weg,  const float* __restrict__ Woe,
                         bf16* __restrict__ WqkvT, bf16* __restrict__ WohT,
                         bf16* __restrict__ WegT,  bf16* __restrict__ WoeT) {
  int i = blockIdx.x * blockDim.x + threadIdx.x;
  if (i < 1536 * 512) { int o = i >> 9, f = i & 511; WqkvT[i] = (bf16)Wqkv[f * 1536 + o]; return; }
  i -= 1536 * 512;
  if (i < 512 * 512)  { int o = i >> 9, f = i & 511; WohT[i]  = (bf16)Woh[f * 512 + o];   return; }
  i -= 512 * 512;
  if (i < 32 * 64)    { int c = i >> 6, k = i & 63;  WegT[i]  = (bf16)Weg[k * 32 + c];    return; }
  i -= 32 * 64;
  if (i < 64 * 16)    { int c = i >> 4, k = i & 15;  WoeT[i]  = (bf16)Woe[k * 64 + c];    return; }
}

// =====================================================================
// Kernel 1: LayerNorm(h) + QKV GEMM (bf16 WMMA), scatter into WMMA layouts
//   Qs  [B][H][N][D] (SCALE pre-applied), Kst [B][N][H][D], Vst [B][H][D][N]
// =====================================================================
__global__ __launch_bounds__(256) void egt_qkv(
    const float* __restrict__ hIn, const float* __restrict__ lnHg, const float* __restrict__ lnHb,
    const float* __restrict__ bqkv, const bf16* __restrict__ WqkvT,
    bf16* __restrict__ Qs, bf16* __restrict__ Kst, bf16* __restrict__ Vst) {
  __shared__ bf16 hLn[16 * 528];
  __shared__ float redS[256], redQ[256], meanS[16], rstdS[16];

  const int tid = threadIdx.x;
  const int b = blockIdx.x >> 5, lt = blockIdx.x & 31, l0 = lt * 16;
  const int r = tid >> 4, seg = tid & 15;

  // ---- LayerNorm over 512 (16 threads per row, 32 elems each); h is streaming -> NT loads ----
  const f4v* hp = (const f4v*)(hIn + ((size_t)(b * NN + l0 + r)) * NN + seg * 32);
  f4v xv[8];
  float s = 0.f, q = 0.f;
#pragma unroll
  for (int i = 0; i < 8; ++i) {
    f4v v = __builtin_nontemporal_load(hp + i); xv[i] = v;
    s += v[0] + v[1] + v[2] + v[3];
    q += v[0]*v[0] + v[1]*v[1] + v[2]*v[2] + v[3]*v[3];
  }
  redS[tid] = s; redQ[tid] = q;
  __syncthreads();
  if (tid < 16) {
    float ss = 0.f, qq = 0.f;
#pragma unroll
    for (int i = 0; i < 16; ++i) { ss += redS[tid * 16 + i]; qq += redQ[tid * 16 + i]; }
    float m = ss * (1.f / 512.f);
    float v = qq * (1.f / 512.f) - m * m;
    meanS[tid] = m; rstdS[tid] = rsqrtf(v + 1e-5f);
  }
  __syncthreads();
  {
    const float m = meanS[r], rs = rstdS[r];
    const f4v* gp = (const f4v*)(lnHg + seg * 32);
    const f4v* bp = (const f4v*)(lnHb + seg * 32);
    v8bf* dst = (v8bf*)(hLn + r * 528 + seg * 32);
#pragma unroll
    for (int g = 0; g < 4; ++g) {
      f4v a = xv[2 * g], c = xv[2 * g + 1];
      f4v ga = gp[2 * g], gb = gp[2 * g + 1], ba = bp[2 * g], bb = bp[2 * g + 1];
      v8bf t;
#pragma unroll
      for (int i = 0; i < 4; ++i) {
        t[i]     = (bf16)((a[i] - m) * rs * ga[i] + ba[i]);
        t[i + 4] = (bf16)((c[i] - m) * rs * gb[i] + bb[i]);
      }
      dst[g] = t;
    }
  }
  __syncthreads();

  // ---- GEMM 16 x 1536 (K=512): 96 N-tiles, 12 per wave, 16 K-steps ----
  const int lane = tid & 31, wv = tid >> 5, ln15 = lane & 15;
  const bool hi = lane >= 16;
  const int k0 = hi ? 16 : 0;
  v8f acc[12];
#pragma unroll
  for (int i = 0; i < 12; ++i) acc[i] = zf();

  for (int k = 0; k < 16; ++k) {
    const bf16* ap = hLn + ln15 * 528 + 32 * k + k0;
    v16bf a = cat8(*(const v8bf*)ap, *(const v8bf*)(ap + 8));
#pragma unroll
    for (int i = 0; i < 12; ++i) {
      int o = (wv * 12 + i) * 16 + ln15;
      v16bf bfr = *(const v16bf*)(WqkvT + (size_t)o * 512 + 32 * k + (hi ? 16 : 0));
      acc[i] = wmma_bf16(a, bfr, acc[i]);
    }
  }
#pragma unroll
  for (int i = 0; i < 12; ++i) {
    int o = (wv * 12 + i) * 16 + ln15;
    float bq = bqkv[o];
#pragma unroll
    for (int j = 0; j < 8; ++j) {
      int l = l0 + j + (hi ? 8 : 0);
      float v = acc[i][j] + bq;
      if (o < 512) {
        int d = o >> 4, hd = o & 15;
        Qs[(((size_t)b * HH + hd) * NN + l) * DD + d] = (bf16)(v * QSCALE);
      } else if (o < 1024) {
        int f = o - 512, d = f >> 4, hd = f & 15;
        Kst[(((size_t)b * NN + l) * HH + hd) * DD + d] = (bf16)v;
      } else {
        int f = o - 1024, d = f >> 4, hd = f & 15;
        Vst[(((size_t)b * HH + hd) * DD + d) * NN + l] = (bf16)v;
      }
    }
  }
}

// =====================================================================
// Kernel 2: fused edge pipeline. One WG per (b, 16-row l-tile); 8 waves.
// Streams m in chunks of 16; flash-style online softmax*gates; e_out fused.
// =====================================================================
__global__ __launch_bounds__(256) void egt_main(
    const float* __restrict__ e, const float* __restrict__ maskp,
    const float* __restrict__ lnEg, const float* __restrict__ lnEb,
    const bf16* __restrict__ Qs, const bf16* __restrict__ Kst, const bf16* __restrict__ Vst,
    const bf16* __restrict__ WegT, const bf16* __restrict__ WoeT, const bf16* __restrict__ WohT,
    const float* __restrict__ b_eg, const float* __restrict__ b_oe, const float* __restrict__ b_oh,
    float* __restrict__ hOut, float* __restrict__ eOut) {
  extern __shared__ __align__(128) char smem[];
  bf16*  eLnB  = (bf16*)(smem + OFF_ELN);
  float* EbufF = (float*)(smem + OFF_EBUF);  // E (+bias); aliased as Hf after QK phase
  float* GbufF = (float*)(smem + OFF_GBUF);
  bf16*  HbB   = (bf16*)(smem + OFF_HB);
  bf16*  KchB  = (bf16*)(smem + OFF_KCH);
  bf16*  VchB  = (bf16*)(smem + OFF_VCH);
  bf16*  AbB   = (bf16*)(smem + OFF_AB);
  float* maskF = (float*)(smem + OFF_MASK);
  float* MrowF = (float*)(smem + OFF_MROW);
  float* SrowF = (float*)(smem + OFF_SROW);
  float* DegF  = (float*)(smem + OFF_DEG);
  float* FacF  = (float*)(smem + OFF_FAC);
  float* lnGF  = (float*)(smem + OFF_LNG);
  float* lnBF  = (float*)(smem + OFF_LNB);

  const int tid = threadIdx.x;
  const int b = blockIdx.x >> 5, lt = blockIdx.x & 31, l0 = lt * 16;
  const int lane = tid & 31, wv = tid >> 5, ln15 = lane & 15;
  const bool hi = lane >= 16;
  const int k0 = hi ? 16 : 0;

  // ---- one-time init ----
  MrowF[tid] = -3.0e38f; SrowF[tid] = 0.f; DegF[tid] = 0.f;
  if (tid < 64) { lnGF[tid] = lnEg[tid]; lnBF[tid] = lnEb[tid]; }

  // ---- persistent fragments ----
  v16bf Qfr[2];
#pragma unroll
  for (int hh = 0; hh < 2; ++hh) {
    const bf16* qp = Qs + (((size_t)b * HH + (2 * wv + hh)) * NN + l0 + ln15) * DD + k0;
    Qfr[hh] = cat8(*(const v8bf*)qp, *(const v8bf*)(qp + 8));
  }
  v16bf WegB[2][2];
#pragma unroll
  for (int nt = 0; nt < 2; ++nt)
#pragma unroll
    for (int ks = 0; ks < 2; ++ks) {
      int c = nt * 16 + ln15;
      WegB[nt][ks] = *(const v16bf*)(WegT + c * 64 + 32 * ks + (hi ? 16 : 0));
    }
  v16bf WoeB[4]; float boe[4];
#pragma unroll
  for (int nt = 0; nt < 4; ++nt) {
    int c = nt * 16 + ln15;
    WoeB[nt] = hi ? z16() : *(const v16bf*)(WoeT + c * 16);  // K=16..31 rows are zero
    boe[nt] = b_oe[c];
  }
  const float begE = b_eg[ln15];
  const float begG = b_eg[16 + ln15];

  v8f accAV[2][2];  // [head in wave][d-tile] persistent flash accumulators
#pragma unroll
  for (int i = 0; i < 2; ++i)
#pragma unroll
    for (int j = 0; j < 2; ++j) accAV[i][j] = zf();

  for (int mc = 0; mc < 32; ++mc) {
    const int m0 = mc * 16;
    __syncthreads();

    // ---------- P1: stage K/V (async if available), LN(e) chunk, mask ----------
    {
      // K chunk -> [h][mi][d]; V chunk -> [h][d][mi] : issue first so they overlap LN math
#ifdef EGT_ASYNC_LDS
      {
        int ih = tid & 15, im = tid >> 4;
        const bf16* src = Kst + (((size_t)b * NN + m0 + im) * HH + ih) * DD;
        bf16* dst = KchB + (ih * 16 + im) * 32;
#pragma unroll
        for (int pc = 0; pc < 4; ++pc)
          __builtin_amdgcn_global_load_async_to_lds_b128(glb_v4(src + 8 * pc),
                                                         lds_v4(dst + 8 * pc), 0, 0);
      }
#pragma unroll
      for (int qq = 0; qq < 2; ++qq) {
        int idx = tid + qq * 256;
        int ih = idx >> 5, id = idx & 31;
        const bf16* src = Vst + (((size_t)b * HH + ih) * DD + id) * NN + m0;
        bf16* dst = VchB + (ih * 32 + id) * 16;
        __builtin_amdgcn_global_load_async_to_lds_b128(glb_v4(src),     lds_v4(dst),     0, 0);
        __builtin_amdgcn_global_load_async_to_lds_b128(glb_v4(src + 8), lds_v4(dst + 8), 0, 0);
      }
#else
      {
        int ih = tid & 15, im = tid >> 4;
        const v8bf* src = (const v8bf*)(Kst + (((size_t)b * NN + m0 + im) * HH + ih) * DD);
        v8bf* d2 = (v8bf*)(KchB + (ih * 16 + im) * 32);
        d2[0] = src[0]; d2[1] = src[1]; d2[2] = src[2]; d2[3] = src[3];
      }
#pragma unroll
      for (int qq = 0; qq < 2; ++qq) {
        int idx = tid + qq * 256;
        int ih = idx >> 5, id = idx & 31;
        const v8bf* src = (const v8bf*)(Vst + (((size_t)b * HH + ih) * DD + id) * NN + m0);
        v8bf* d2 = (v8bf*)(VchB + (ih * 32 + id) * 16);
        d2[0] = src[0]; d2[1] = src[1];
      }
#endif
      // LN(e): e is strictly streaming -> non-temporal loads (keep L2 for reused operands)
      const int l = tid >> 4, mi = tid & 15;
      const float* ep = e + (((size_t)b * NN + l0 + l) * NN + (m0 + mi)) * EW;
      const f4v* ep4 = (const f4v*)ep;
      f4v xv[16];
      float s = 0.f, q = 0.f;
#pragma unroll
      for (int i = 0; i < 16; ++i) {
        f4v v = __builtin_nontemporal_load(ep4 + i); xv[i] = v;
        s += v[0] + v[1] + v[2] + v[3];
        q += v[0]*v[0] + v[1]*v[1] + v[2]*v[2] + v[3]*v[3];
      }
      if (mc + 1 < 32) {  // prefetch next chunk (global_prefetch_b8)
        __builtin_prefetch(ep + 1024, 0, 0);
        __builtin_prefetch(ep + 1024 + 32, 0, 0);
      }
      float m = s * (1.f / 64.f);
      float var = q * (1.f / 64.f) - m * m;
      float rs = rsqrtf(var + 1e-5f);
      const f4v* g4 = (const f4v*)lnGF;
      const f4v* b4 = (const f4v*)lnBF;
      v8bf* dst = (v8bf*)(eLnB + tid * 64);
#pragma unroll
      for (int g = 0; g < 8; ++g) {
        f4v a = xv[2 * g], c = xv[2 * g + 1];
        f4v ga = g4[2 * g], gb = g4[2 * g + 1], ba = b4[2 * g], bb = b4[2 * g + 1];
        v8bf t;
#pragma unroll
        for (int i = 0; i < 4; ++i) {
          t[i]     = (bf16)((a[i] - m) * rs * ga[i] + ba[i]);
          t[i + 4] = (bf16)((c[i] - m) * rs * gb[i] + bb[i]);
        }
        dst[g] = t;
      }
      maskF[tid] = __builtin_nontemporal_load(maskp + ((size_t)b * NN + l0 + l) * NN + (m0 + mi));
#ifdef EGT_ASYNC_LDS
      egt_wait_async();   // our wave's async K/V copies landed; barrier publishes across waves
#endif
    }
    __syncthreads();

    // ---------- P2: EG = eLn @ W_eg (WMMA); store E+bias, G+bias ----------
    for (int t = 0; t < 2; ++t) {
      int mt = 2 * wv + t;
      const bf16* ap = eLnB + (16 * mt + ln15) * 64 + k0;
      v16bf a0 = cat8(*(const v8bf*)ap,        *(const v8bf*)(ap + 8));
      v16bf a1 = cat8(*(const v8bf*)(ap + 32), *(const v8bf*)(ap + 40));
      v8f aE = zf(), aG = zf();
      aE = wmma_bf16(a0, WegB[0][0], aE);
      aE = wmma_bf16(a1, WegB[0][1], aE);
      aG = wmma_bf16(a0, WegB[1][0], aG);
      aG = wmma_bf16(a1, WegB[1][1], aG);
      int base = 16 * mt + (hi ? 8 : 0);
#pragma unroll
      for (int j = 0; j < 8; ++j) {
        EbufF[ln15 * 260 + base + j] = aE[j] + begE;
        GbufF[ln15 * 260 + base + j] = aG[j] + begG;
      }
    }
    __syncthreads();

    // ---------- P3: H_hat = Q@K^T + E (E as C operand); store f32 + bf16 ----------
#pragma unroll
    for (int hh = 0; hh < 2; ++hh) {
      int h = 2 * wv + hh;
      v8f c;
#pragma unroll
      for (int j = 0; j < 8; ++j)
        c[j] = EbufF[h * 260 + (j + (hi ? 8 : 0)) * 16 + ln15];
      v16bf bk = *(const v16bf*)(KchB + (h * 16 + ln15) * 32 + (hi ? 16 : 0));
      v8f d = wmma_bf16(Qfr[hh], bk, c);
#pragma unroll
      for (int j = 0; j < 8; ++j) {
        int pr = (j + (hi ? 8 : 0)) * 16 + ln15;
        EbufF[h * 260 + pr] = d[j];        // Hf (aliases Ebuf; same addresses)
        HbB[pr * 16 + h] = (bf16)d[j];     // A-operand source for e_out
      }
    }
    __syncthreads();

    // ---------- P4: online softmax * gates, degree, A_tild ----------
    {
      const int l = tid >> 4, h = tid & 15;
      float sv[16];
      float cm = -3.0e38f;
#pragma unroll
      for (int mi = 0; mi < 16; ++mi) {
        float x = EbufF[h * 260 + l * 16 + mi] + maskF[l * 16 + mi];
        sv[mi] = x; cm = fmaxf(cm, x);
      }
      float Mold = MrowF[tid];             // tid == l*16 + h
      float Mnew = fmaxf(Mold, cm);
      float fac = __expf(Mold - Mnew);
      float P = 0.f, GS = 0.f;
      bf16* ab = AbB + h * 272 + l * 16;
#pragma unroll
      for (int mi = 0; mi < 16; ++mi) {
        float p = __expf(sv[mi] - Mnew); P += p;
        float g = GbufF[h * 260 + l * 16 + mi] + maskF[l * 16 + mi];
        float gate = 1.f / (1.f + __expf(-g));
        GS += gate;
        ab[mi] = (bf16)(p * gate);
      }
      SrowF[tid] = SrowF[tid] * fac + P;
      DegF[tid] += GS;
      MrowF[tid] = Mnew;
      FacF[tid] = fac;
    }
    __syncthreads();

    // ---------- P5a: rescale + A_tild @ V (WMMA, K padded to 32) ----------
#pragma unroll
    for (int hh = 0; hh < 2; ++hh) {
      int h = 2 * wv + hh;
      float fj[8];
#pragma unroll
      for (int j = 0; j < 8; ++j) fj[j] = FacF[(j + (hi ? 8 : 0)) * 16 + h];
#pragma unroll
      for (int t = 0; t < 2; ++t)
#pragma unroll
        for (int j = 0; j < 8; ++j) accAV[hh][t][j] *= fj[j];
      const bf16* ap = AbB + h * 272 + ln15 * 16;
      v16bf a = hi ? z16() : cat8(*(const v8bf*)ap, *(const v8bf*)(ap + 8));      // K=16..31 zero
#pragma unroll
      for (int t = 0; t < 2; ++t) {
        v16bf bv = hi ? z16()
                      : *(const v16bf*)(VchB + (h * 32 + 16 * t + ln15) * 16);
        accAV[hh][t] = wmma_bf16(a, bv, accAV[hh][t]);
      }
    }
    // ---------- P5b: e_out = H_hat @ W_oe + b_oe (WMMA, NT store direct) ----------
    for (int t = 0; t < 2; ++t) {
      int mt = 2 * wv + t;
      const bf16* ap = HbB + (16 * mt + ln15) * 16;
      v16bf a = hi ? z16() : cat8(*(const v8bf*)ap, *(const v8bf*)(ap + 8));      // heads padded to K=32
#pragma unroll
      for (int nt = 0; nt < 4; ++nt) {
        v8f dfr = wmma_bf16(a, WoeB[nt], zf());
        int cI = nt * 16 + ln15;
#pragma unroll
        for (int j = 0; j < 8; ++j) {
          int pr = 16 * mt + j + (hi ? 8 : 0);
          int l = pr >> 4, mi = pr & 15;
          __builtin_nontemporal_store(
              dfr[j] + boe[nt],
              &eOut[(((size_t)b * NN + l0 + l) * NN + (m0 + mi)) * EW + cI]);
        }
      }
    }
  }
  __syncthreads();

  // ---------- Final: scale by log1p(deg)/denom, then h_out = Vatt @ W_oh ----------
  {
    bf16* vat = (bf16*)(smem + OFF_ELN);  // [16][528] bf16, reuses eLn region
#pragma unroll
    for (int hh = 0; hh < 2; ++hh) {
      int h = 2 * wv + hh;
      float sc[8];
#pragma unroll
      for (int j = 0; j < 8; ++j) {
        int rr = (j + (hi ? 8 : 0)) * 16 + h;
        sc[j] = log1pf(DegF[rr]) / SrowF[rr];
      }
#pragma unroll
      for (int t = 0; t < 2; ++t)
#pragma unroll
        for (int j = 0; j < 8; ++j) {
          int l = j + (hi ? 8 : 0);
          int f = (16 * t + ln15) * 16 + h;  // feature = d*16 + h
          vat[l * 528 + f] = (bf16)(accAV[hh][t][j] * sc[j]);
        }
    }
  }
  __syncthreads();
  {
    const bf16* vat = (const bf16*)(smem + OFF_ELN);
    v8f acc4[4];
#pragma unroll
    for (int i = 0; i < 4; ++i) acc4[i] = zf();
    for (int k = 0; k < 16; ++k) {
      const bf16* ap = vat + ln15 * 528 + 32 * k + k0;
      v16bf a = cat8(*(const v8bf*)ap, *(const v8bf*)(ap + 8));
#pragma unroll
      for (int i = 0; i < 4; ++i) {
        int o = (wv * 4 + i) * 16 + ln15;
        v16bf bfr = *(const v16bf*)(WohT + (size_t)o * 512 + 32 * k + (hi ? 16 : 0));
        acc4[i] = wmma_bf16(a, bfr, acc4[i]);
      }
    }
#pragma unroll
    for (int i = 0; i < 4; ++i) {
      int o = (wv * 4 + i) * 16 + ln15;
      float bo = b_oh[o];
#pragma unroll
      for (int j = 0; j < 8; ++j) {
        int l = j + (hi ? 8 : 0);
        __builtin_nontemporal_store(acc4[i][j] + bo,
                                    &hOut[((size_t)b * NN + l0 + l) * NN + o]);
      }
    }
  }
}

// =====================================================================
// Host launcher
// =====================================================================
extern "C" void kernel_launch(void* const* d_in, const int* in_sizes, int n_in,
                              void* d_out, int out_size, void* d_ws, size_t ws_size,
                              hipStream_t stream) {
  (void)in_sizes; (void)n_in; (void)out_size; (void)ws_size;
  const float* h      = (const float*)d_in[0];
  const float* e      = (const float*)d_in[1];
  const float* mask   = (const float*)d_in[2];
  const float* ln_h_g = (const float*)d_in[3];
  const float* ln_h_b = (const float*)d_in[4];
  const float* ln_e_g = (const float*)d_in[5];
  const float* ln_e_b = (const float*)d_in[6];
  const float* W_qkv  = (const float*)d_in[7];
  const float* b_qkv  = (const float*)d_in[8];
  const float* W_eg   = (const float*)d_in[9];
  const float* b_eg   = (const float*)d_in[10];
  const float* W_oh   = (const float*)d_in[11];
  const float* b_oh   = (const float*)d_in[12];
  const float* W_oe   = (const float*)d_in[13];
  const float* b_oe   = (const float*)d_in[14];

  char* ws = (char*)d_ws;
  bf16* WqkvT = (bf16*)(ws + WS_WQKVT);
  bf16* WohT  = (bf16*)(ws + WS_WOHT);
  bf16* WegT  = (bf16*)(ws + WS_WEGT);
  bf16* WoeT  = (bf16*)(ws + WS_WOET);
  bf16* Qs    = (bf16*)(ws + WS_QS);
  bf16* Kst   = (bf16*)(ws + WS_KST);
  bf16* Vst   = (bf16*)(ws + WS_VST);

  float* hOut = (float*)d_out;
  float* eOut = hOut + (size_t)BB * NN * NN;

  const int prepN = 1536 * 512 + 512 * 512 + 32 * 64 + 64 * 16;
  egt_prep<<<(prepN + 255) / 256, 256, 0, stream>>>(W_qkv, W_oh, W_eg, W_oe,
                                                    WqkvT, WohT, WegT, WoeT);
  egt_qkv<<<BB * (NN / 16), 256, 0, stream>>>(h, ln_h_g, ln_h_b, b_qkv, WqkvT, Qs, Kst, Vst);
  egt_main<<<BB * (NN / 16), 256, SMEM_SZ, stream>>>(e, mask, ln_e_g, ln_e_b,
                                                     Qs, Kst, Vst, WegT, WoeT, WohT,
                                                     b_eg, b_oe, b_oh, hOut, eOut);
}
